// BaichuanAttention_5626407157952
// MI455X (gfx1250) — matrix operand order repaired and checked by reading, not compiled, not run on hardware
//
#include <hip/hip_runtime.h>

#define SQ   2048
#define DIM  4096
#define NH   32
#define HD   128
#define HIST 1024
#define TT   3072   // HIST + SQ

typedef __attribute__((ext_vector_type(16))) __bf16 v16bf;
typedef __attribute__((ext_vector_type(8)))  float  v8f;
typedef int v4i __attribute__((__vector_size__(16)));   // matches builtin param type

#if defined(__has_builtin)
#  if __has_builtin(__builtin_amdgcn_global_load_async_to_lds_b128)
#    define USE_ASYNC_LDS 1
#  endif
#endif
#ifndef USE_ASYNC_LDS
#  define USE_ASYNC_LDS 0
#endif

static __device__ __forceinline__ unsigned short f2bf(float x) {
  unsigned int u = __float_as_uint(x);
  u += 0x7fffu + ((u >> 16) & 1u);          // round-to-nearest-even
  return (unsigned short)(u >> 16);
}

// 16-byte global -> LDS copy; async (ASYNCcnt) when the toolchain exposes it.
static __device__ __forceinline__ void copy_b128_to_lds(const unsigned short* g,
                                                        unsigned short* l) {
#if USE_ASYNC_LDS
  __builtin_amdgcn_global_load_async_to_lds_b128(
      (__attribute__((address_space(1))) v4i*)(unsigned long long)g,
      (__attribute__((address_space(3))) v4i*)(unsigned int)(unsigned long long)l,
      0, 0);
#else
  *(uint4*)l = *(const uint4*)g;
#endif
}

static __device__ __forceinline__ void async_wait0() {
#if USE_ASYNC_LDS
#  if __has_builtin(__builtin_amdgcn_s_wait_asynccnt)
  __builtin_amdgcn_s_wait_asynccnt(0);
#  else
  asm volatile("s_wait_asynccnt 0x0" ::: "memory");
#  endif
#endif
}

union FragU { v16bf v; uint4 u[2]; };

// A-matrix 16x32 bf16 layout: lane m=l&15, chunks K=[kb..kb+7] and [kb+16..kb+23], kb=(l&16)?8:0
static __device__ __forceinline__ v16bf load_a_frag(const unsigned short* p, int ldk, int lane) {
  const unsigned short* r = p + (lane & 15) * ldk + ((lane & 16) ? 8 : 0);
  FragU f;
  f.u[0] = *(const uint4*)(r);
  f.u[1] = *(const uint4*)(r + 16);
  return f.v;
}

// B-matrix 32x16 bf16 layout: lane n=l&15 holds column n, chunks K=[kb..kb+7] and [kb+16..kb+23], kb=(l&16)?8:0
// (p points at N x K row-major storage, i.e. B^T)
static __device__ __forceinline__ v16bf load_b_frag(const unsigned short* p, int ldk, int lane) {
  const unsigned short* r = p + (lane & 15) * ldk + ((lane & 16) ? 8 : 0);
  FragU f;
  f.u[0] = *(const uint4*)(r);
  f.u[1] = *(const uint4*)(r + 16);
  return f.v;
}

#define WMMA_BF16(a, b, c) \
  __builtin_amdgcn_wmma_f32_16x16x32_bf16(false, (a), false, (b), (short)0, (c), false, false)

// ---------------------------------------------------------------------------
// Kernel 1: convert past K/V fp32 -> bf16 into [H][T][hd] caches (first HIST rows)
// ---------------------------------------------------------------------------
__global__ __launch_bounds__(256) void k_convert_past(
    const float* __restrict__ pk, const float* __restrict__ pv,
    unsigned short* __restrict__ kall, unsigned short* __restrict__ vall) {
  int c = blockIdx.x * 256 + threadIdx.x;   // chunk of 4 elements
  int e = c * 4;
  int h   = e / (HIST * HD);
  int rem = e - h * (HIST * HD);
  float4 a = *(const float4*)(pk + e);
  float4 b = *(const float4*)(pv + e);
  ushort4 ua = { f2bf(a.x), f2bf(a.y), f2bf(a.z), f2bf(a.w) };
  ushort4 ub = { f2bf(b.x), f2bf(b.y), f2bf(b.z), f2bf(b.w) };
  int dst = h * (TT * HD) + rem;
  *(ushort4*)(kall + dst) = ua;
  *(ushort4*)(vall + dst) = ub;
}

// ---------------------------------------------------------------------------
// Kernel 2: QKV GEMM  C[m,n] = sum_k X[m,k] * Wpack[n,k]
//   M=2048, N=12288, K=4096; fp32 operands converted to bf16 while staging
//   (conversion needs the VGPR path, so staging uses regular stores here).
//   Epilogue routes n<4096 -> Q, <8192 -> K cache, else V cache (bf16).
// ---------------------------------------------------------------------------
__global__ __launch_bounds__(256) void k_gemm_qkv(
    const float* __restrict__ A, const float* __restrict__ W,
    unsigned short* __restrict__ qb, unsigned short* __restrict__ kall,
    unsigned short* __restrict__ vall) {
  __shared__ __align__(16) unsigned short As[128 * 32];
  __shared__ __align__(16) unsigned short Bs[128 * 32];
  const int tid  = threadIdx.x;
  const int lane = tid & 31;
  const int wave = tid >> 5;
  const int wm = (wave & 3) * 32;
  const int wn = (wave >> 2) * 64;
  const int m0 = blockIdx.y * 128;
  const int n0 = blockIdx.x * 128;

  v8f zero8 = {0.f,0.f,0.f,0.f,0.f,0.f,0.f,0.f};
  v8f acc[2][4];
  #pragma unroll
  for (int i = 0; i < 2; ++i)
    #pragma unroll
    for (int j = 0; j < 4; ++j) acc[i][j] = zero8;

  for (int k0 = 0; k0 < DIM; k0 += 32) {
    #pragma unroll
    for (int i = 0; i < 4; ++i) {                 // stage A tile (fp32 -> bf16)
      int c = tid + i * 256;
      int row = c >> 3, col = (c & 7) * 4;
      float4 f = *(const float4*)(A + (size_t)(m0 + row) * DIM + k0 + col);
      ushort4 h4 = { f2bf(f.x), f2bf(f.y), f2bf(f.z), f2bf(f.w) };
      *(ushort4*)(As + row * 32 + col) = h4;
    }
    #pragma unroll
    for (int i = 0; i < 4; ++i) {                 // stage W tile (N x K)
      int c = tid + i * 256;
      int row = c >> 3, col = (c & 7) * 4;
      float4 f = *(const float4*)(W + (size_t)(n0 + row) * DIM + k0 + col);
      ushort4 h4 = { f2bf(f.x), f2bf(f.y), f2bf(f.z), f2bf(f.w) };
      *(ushort4*)(Bs + row * 32 + col) = h4;
    }
    __syncthreads();
    v16bf af[2], bf[4];
    #pragma unroll
    for (int mt = 0; mt < 2; ++mt) af[mt] = load_a_frag(As + (wm + mt * 16) * 32, 32, lane);
    #pragma unroll
    for (int nt = 0; nt < 4; ++nt) bf[nt] = load_b_frag(Bs + (wn + nt * 16) * 32, 32, lane);
    #pragma unroll
    for (int mt = 0; mt < 2; ++mt)
      #pragma unroll
      for (int nt = 0; nt < 4; ++nt)
        acc[mt][nt] = WMMA_BF16(af[mt], bf[nt], acc[mt][nt]);
    __syncthreads();
  }

  #pragma unroll
  for (int mt = 0; mt < 2; ++mt)
    #pragma unroll
    for (int nt = 0; nt < 4; ++nt)
      #pragma unroll
      for (int r = 0; r < 8; ++r) {
        int m = m0 + wm + mt * 16 + r + ((lane >> 4) << 3);
        int n = n0 + wn + nt * 16 + (lane & 15);
        unsigned short bv = f2bf(acc[mt][nt][r]);
        int seg = n >> 12;
        int f = n & 4095;
        int h = f >> 7, d = f & 127;
        if (seg == 0)      qb[((size_t)h * SQ + m) * HD + d] = bv;
        else if (seg == 1) kall[((size_t)h * TT + HIST + m) * HD + d] = bv;
        else               vall[((size_t)h * TT + HIST + m) * HD + d] = bv;
      }
}

// ---------------------------------------------------------------------------
// Kernel 3: flash attention, one block = (head, 128 queries), 8 waves x 16 q.
//   K tile staged with async global->LDS (ASYNCcnt); next K/V tiles
//   prefetched; ALiBi bias + causal mask in f32 accumulators; online
//   softmax; P reuses the K LDS region after a barrier (LDS = 64 KB).
// ---------------------------------------------------------------------------
__global__ __launch_bounds__(256) void k_attn(
    const unsigned short* __restrict__ qb,
    const unsigned short* __restrict__ kall,
    const unsigned short* __restrict__ vall,
    unsigned short* __restrict__ ctx) {
  __shared__ __align__(16) unsigned short Ks[128 * 128];  // [key][d]; later P [wave][16][128]
  __shared__ __align__(16) unsigned short Vt[128 * 128];  // [d][key]
  const int tid  = threadIdx.x;
  const int lane = tid & 31;
  const int wave = tid >> 5;
  const int m0 = blockIdx.x * 128;
  const int h  = blockIdx.y;
  const float scale = 0.08838834764831845f;             // 1/sqrt(128)
  const float slope = exp2f(-0.25f * (float)(h + 1));   // ALiBi, H=32 power of 2

  // Q fragments, loaded once directly from global (A-layout matches row-major [q][d])
  v16bf aq[4];
  const unsigned short* qbase = qb + ((size_t)h * SQ + m0 + wave * 16) * HD;
  #pragma unroll
  for (int kk = 0; kk < 4; ++kk) aq[kk] = load_a_frag(qbase + kk * 32, HD, lane);

  v8f zero8 = {0.f,0.f,0.f,0.f,0.f,0.f,0.f,0.f};
  v8f oacc[8];
  float mrun[8], lrun[8];
  #pragma unroll
  for (int i = 0; i < 8; ++i) { oacc[i] = zero8; mrun[i] = -1e30f; lrun[i] = 0.f; }

  const int kt_end = HIST + m0 + 128;                   // causal upper bound for this tile
  for (int kt = 0; kt < kt_end; kt += 128) {
    const unsigned short* gk = kall + ((size_t)h * TT + kt) * HD;
    const unsigned short* gv = vall + ((size_t)h * TT + kt) * HD;
    #pragma unroll
    for (int i = 0; i < 8; ++i) {                       // K tile: contiguous 32KB copy
      int e = tid + i * 256;
      copy_b128_to_lds(gk + e * 8, Ks + e * 8);
    }
    #pragma unroll
    for (int i = 0; i < 8; ++i) {                       // V tile: transpose into [d][key]
      int e = tid + i * 256;
      int row = e >> 4, col = (e & 15) * 8;
      uint4 u = *(const uint4*)(gv + row * HD + col);
      const unsigned short* s = (const unsigned short*)&u;
      #pragma unroll
      for (int j = 0; j < 8; ++j) Vt[(col + j) * 128 + row] = s[j];
    }
    if (kt + 128 < kt_end) {                            // prefetch next K/V tile
      __builtin_prefetch(gk + 128 * HD + tid * 64, 0, 1);
      __builtin_prefetch(gv + 128 * HD + tid * 64, 0, 1);
    }
    async_wait0();                                      // own async K writes landed
    __syncthreads();

    v8f sacc[8];
    #pragma unroll
    for (int i = 0; i < 8; ++i) sacc[i] = zero8;
    #pragma unroll
    for (int nt = 0; nt < 8; ++nt)
      #pragma unroll
      for (int kk = 0; kk < 4; ++kk) {
        v16bf bk = load_b_frag(Ks + (nt * 16) * HD + kk * 32, HD, lane);
        sacc[nt] = WMMA_BF16(aq[kk], bk, sacc[nt]);
      }

    const int qrow = m0 + wave * 16 + ((lane >> 4) << 3);   // + r  -> local query row
    const int kcol = kt + (lane & 15);                      // + nt*16 -> key position
    #pragma unroll
    for (int r = 0; r < 8; ++r) {
      int qp = HIST + qrow + r;
      float mx = -1e30f;
      #pragma unroll
      for (int nt = 0; nt < 8; ++nt) {
        float rel = (float)(kcol + nt * 16 - qp);
        float sv  = sacc[nt][r];
        sv = (rel <= 0.0f) ? (sv * scale + slope * rel) : -1e30f;
        sacc[nt][r] = sv;
        mx = fmaxf(mx, sv);
      }
      #pragma unroll
      for (int off = 8; off >= 1; off >>= 1)                // reduce over 16-lane row group
        mx = fmaxf(mx, __shfl_xor(mx, off, 32));
      float mnew  = fmaxf(mrun[r], mx);
      float alpha = __expf(mrun[r] - mnew);
      mrun[r] = mnew;
      float ps = 0.f;
      #pragma unroll
      for (int nt = 0; nt < 8; ++nt) {
        float p = __expf(sacc[nt][r] - mnew);
        sacc[nt][r] = p;
        ps += p;
      }
      #pragma unroll
      for (int off = 8; off >= 1; off >>= 1)
        ps += __shfl_xor(ps, off, 32);
      lrun[r] = lrun[r] * alpha + ps;
      #pragma unroll
      for (int dt = 0; dt < 8; ++dt) oacc[dt][r] *= alpha;
    }
    __syncthreads();                                        // all waves done reading Ks

    unsigned short* Ps = Ks + wave * (16 * 128);            // per-wave P scratch
    #pragma unroll
    for (int nt = 0; nt < 8; ++nt)
      #pragma unroll
      for (int r = 0; r < 8; ++r)
        Ps[(r + ((lane >> 4) << 3)) * 128 + nt * 16 + (lane & 15)] = f2bf(sacc[nt][r]);

    v16bf pa[4];
    #pragma unroll
    for (int kk = 0; kk < 4; ++kk) pa[kk] = load_a_frag(Ps + kk * 32, 128, lane);
    #pragma unroll
    for (int dt = 0; dt < 8; ++dt)
      #pragma unroll
      for (int kk = 0; kk < 4; ++kk) {
        v16bf bv = load_b_frag(Vt + (dt * 16) * 128 + kk * 32, 128, lane);
        oacc[dt] = WMMA_BF16(pa[kk], bv, oacc[dt]);
      }
    __syncthreads();
  }

  #pragma unroll
  for (int dt = 0; dt < 8; ++dt)
    #pragma unroll
    for (int r = 0; r < 8; ++r) {
      int m = m0 + wave * 16 + r + ((lane >> 4) << 3);
      int d = dt * 16 + (lane & 15);
      float v = oacc[dt][r] / lrun[r];
      ctx[(size_t)m * DIM + h * HD + d] = f2bf(v);          // [s][h*hd+d] = [S, D]
    }
}

// ---------------------------------------------------------------------------
// Kernel 4: output projection  out[m,n] = sum_k ctx[m,k] * Wo[n,k]  (fp32 out)
//   A tile (already bf16) staged with async global->LDS.
// ---------------------------------------------------------------------------
__global__ __launch_bounds__(256) void k_gemm_out(
    const unsigned short* __restrict__ A, const float* __restrict__ W,
    float* __restrict__ out) {
  __shared__ __align__(16) unsigned short As[128 * 32];
  __shared__ __align__(16) unsigned short Bs[128 * 32];
  const int tid  = threadIdx.x;
  const int lane = tid & 31;
  const int wave = tid >> 5;
  const int wm = (wave & 3) * 32;
  const int wn = (wave >> 2) * 64;
  const int m0 = blockIdx.y * 128;
  const int n0 = blockIdx.x * 128;

  v8f zero8 = {0.f,0.f,0.f,0.f,0.f,0.f,0.f,0.f};
  v8f acc[2][4];
  #pragma unroll
  for (int i = 0; i < 2; ++i)
    #pragma unroll
    for (int j = 0; j < 4; ++j) acc[i][j] = zero8;

  for (int k0 = 0; k0 < DIM; k0 += 32) {
    #pragma unroll
    for (int i = 0; i < 2; ++i) {                 // stage A tile (bf16, pure copy)
      int c = tid + i * 256;
      int row = c >> 2, col = (c & 3) * 8;
      copy_b128_to_lds(A + (size_t)(m0 + row) * DIM + k0 + col, As + row * 32 + col);
    }
    #pragma unroll
    for (int i = 0; i < 4; ++i) {                 // stage W tile (fp32 -> bf16)
      int c = tid + i * 256;
      int row = c >> 3, col = (c & 7) * 4;
      float4 f = *(const float4*)(W + (size_t)(n0 + row) * DIM + k0 + col);
      ushort4 h4 = { f2bf(f.x), f2bf(f.y), f2bf(f.z), f2bf(f.w) };
      *(ushort4*)(Bs + row * 32 + col) = h4;
    }
    async_wait0();
    __syncthreads();
    v16bf af[2], bf[4];
    #pragma unroll
    for (int mt = 0; mt < 2; ++mt) af[mt] = load_a_frag(As + (wm + mt * 16) * 32, 32, lane);
    #pragma unroll
    for (int nt = 0; nt < 4; ++nt) bf[nt] = load_b_frag(Bs + (wn + nt * 16) * 32, 32, lane);
    #pragma unroll
    for (int mt = 0; mt < 2; ++mt)
      #pragma unroll
      for (int nt = 0; nt < 4; ++nt)
        acc[mt][nt] = WMMA_BF16(af[mt], bf[nt], acc[mt][nt]);
    __syncthreads();
  }

  #pragma unroll
  for (int mt = 0; mt < 2; ++mt)
    #pragma unroll
    for (int nt = 0; nt < 4; ++nt)
      #pragma unroll
      for (int r = 0; r < 8; ++r) {
        int m = m0 + wm + mt * 16 + r + ((lane >> 4) << 3);
        int n = n0 + wn + nt * 16 + (lane & 15);
        out[(size_t)m * DIM + n] = acc[mt][nt][r];
      }
}

// ---------------------------------------------------------------------------
extern "C" void kernel_launch(void* const* d_in, const int* in_sizes, int n_in,
                              void* d_out, int out_size, void* d_ws, size_t ws_size,
                              hipStream_t stream) {
  const float* hs = (const float*)d_in[0];   // [1, 2048, 4096]
  const float* pk = (const float*)d_in[1];   // [1, 32, 1024, 128]
  const float* pv = (const float*)d_in[2];   // [1, 32, 1024, 128]
  const float* wp = (const float*)d_in[3];   // [12288, 4096]
  const float* wo = (const float*)d_in[4];   // [4096, 4096]
  float* out = (float*)d_out;                // [1, 2048, 4096]

  unsigned char* ws = (unsigned char*)d_ws;
  size_t off = 0;
  unsigned short* qb   = (unsigned short*)(ws + off); off += (size_t)NH * SQ * HD * 2;
  unsigned short* kall = (unsigned short*)(ws + off); off += (size_t)NH * TT * HD * 2;
  unsigned short* vall = (unsigned short*)(ws + off); off += (size_t)NH * TT * HD * 2;
  unsigned short* ctx  = (unsigned short*)(ws + off); off += (size_t)SQ * DIM * 2;
  (void)ws_size; (void)in_sizes; (void)n_in; (void)out_size;

  k_convert_past<<<dim3((NH * HIST * HD) / 4 / 256), 256, 0, stream>>>(pk, pv, kall, vall);
  k_gemm_qkv<<<dim3((3 * DIM) / 128, SQ / 128), 256, 0, stream>>>(hs, wp, qb, kall, vall);
  k_attn<<<dim3(SQ / 128, NH), 256, 0, stream>>>(qb, kall, vall, ctx);
  k_gemm_out<<<dim3(DIM / 128, SQ / 128), 256, 0, stream>>>(ctx, wo, out);
}
